// RoutedExpertsOnlyDecoderLayer_18322330485348
// MI455X (gfx1250) — compile-verified, matrix operand order checked by reading, not run on hardware
//
#include <hip/hip_runtime.h>
#include <stdint.h>

// Problem constants (B,S,D,M,E,K = 4,2048,2048,4096,8,2)
#define TOK   8192      // B*S tokens
#define DDIM  2048
#define MDIM  4096
#define NEXP  8
#define TCH   2048      // token chunk (bounds h workspace)

// GEMM tiling: 256 threads = 8 waves (wave32). Block tile 128x64, K-step 32.
// Waves 4(row groups) x 2(col groups); each wave = 32x32 = 2x2 WMMA 16x16x32.
#define BM  128
#define BN  64
#define BK  32
#define LDW (BK + 8)    // padded LDS row stride (elems): 80B -> conflict-free b128

typedef __attribute__((ext_vector_type(16))) __bf16 v16bf;
typedef __attribute__((ext_vector_type(8)))  __bf16 v8bf;
typedef __attribute__((ext_vector_type(8)))  float  v8f;
typedef __attribute__((ext_vector_type(4)))  int    v4i;

static __device__ __forceinline__ __bf16 f32_bf16(float f) {
  union { float f; uint32_t u; } v; v.f = f;
  uint32_t r = v.u + 0x7FFFu + ((v.u >> 16) & 1u);   // round-to-nearest-even
  union { uint16_t u; __bf16 b; } o; o.u = (uint16_t)(r >> 16);
  return o.b;
}

static __device__ __forceinline__ v16bf cat16(v8bf lo, v8bf hi) {
  return __builtin_shufflevector(lo, hi, 0,1,2,3,4,5,6,7,8,9,10,11,12,13,14,15);
}

// ---- 16B global->LDS tile copy: async-to-LDS when the toolchain has it ----
#if defined(__HIP_DEVICE_COMPILE__) && defined(__has_builtin)
#if __has_builtin(__builtin_amdgcn_global_load_async_to_lds_b128) && \
    __has_builtin(__builtin_amdgcn_s_wait_asynccnt)
#define USE_ASYNC_COPY 1
#endif
#endif

#if USE_ASYNC_COPY
typedef __attribute__((address_space(1))) v4i g_v4i;   // global int4
typedef __attribute__((address_space(3))) v4i l_v4i;   // LDS int4
#endif

static __device__ __forceinline__ void cp8(const __bf16* g, __bf16* l) {
#if USE_ASYNC_COPY
  // generic LDS pointer low 32 bits == LDS byte offset (flat aperture rule)
  __builtin_amdgcn_global_load_async_to_lds_b128(
      (g_v4i*)(uintptr_t)g,
      (l_v4i*)(uint32_t)(uintptr_t)l,
      0, 0);
#else
  *(v8bf*)l = *(const v8bf*)g;
#endif
}
static __device__ __forceinline__ void copy_wait() {
#if USE_ASYNC_COPY
  __builtin_amdgcn_s_wait_asynccnt(0);
#endif
}

// -------------------------------------------------------------------------
// Router: one wave per token; logits = x[t,:] @ gate_w, wave32 butterfly
// reduce, top-2 + softmax, write dense combine weights cw[t,0..7].
// -------------------------------------------------------------------------
__global__ __launch_bounds__(256)
void moe_router(const float* __restrict__ x,
                const float* __restrict__ gw,
                float* __restrict__ cw) {
  const int lane = threadIdx.x & 31;
  const int wid  = threadIdx.x >> 5;
  const int t    = blockIdx.x * 8 + wid;

  float acc[NEXP];
#pragma unroll
  for (int e = 0; e < NEXP; ++e) acc[e] = 0.f;

  const float* xr = x + (size_t)t * DDIM;
  for (int d = lane; d < DDIM; d += 32) {
    float xv = xr[d];
    const float* g = gw + (size_t)d * NEXP;
#pragma unroll
    for (int e = 0; e < NEXP; ++e) acc[e] += xv * g[e];
  }
#pragma unroll
  for (int off = 16; off; off >>= 1) {
#pragma unroll
    for (int e = 0; e < NEXP; ++e) acc[e] += __shfl_xor(acc[e], off, 32);
  }
  int i1 = 0; float l1 = acc[0];
#pragma unroll
  for (int e = 1; e < NEXP; ++e) if (acc[e] > l1) { l1 = acc[e]; i1 = e; }
  int i2 = -1; float l2 = -3.4e38f;
#pragma unroll
  for (int e = 0; e < NEXP; ++e) if (e != i1 && acc[e] > l2) { l2 = acc[e]; i2 = e; }
  float e2  = __expf(l2 - l1);
  float inv = 1.f / (1.f + e2);
  if (lane < NEXP) {
    float w = (lane == i1) ? inv : ((lane == i2) ? e2 * inv : 0.f);
    cw[(size_t)t * NEXP + lane] = w;
  }
}

// -------------------------------------------------------------------------
// Elementwise fp32 -> bf16 (activations), 8 elems/thread, vector stores.
// -------------------------------------------------------------------------
__global__ __launch_bounds__(256)
void cvt_bf16(const float* __restrict__ in, __bf16* __restrict__ out) {
  size_t i = ((size_t)blockIdx.x * 256 + threadIdx.x) * 8;
  v8bf r;
#pragma unroll
  for (int u = 0; u < 8; ++u) r[u] = f32_bf16(in[i + u]);
  *(v8bf*)(out + i) = r;
}

// -------------------------------------------------------------------------
// Tiled transpose + convert: in [R][C] fp32 -> out [C][R] bf16.
// 64x64 tiles through padded LDS; coalesced fp32 reads, vector bf16 writes.
// -------------------------------------------------------------------------
__global__ __launch_bounds__(256)
void transpose_cvt(const float* __restrict__ in, __bf16* __restrict__ out,
                   int R, int C) {
  __shared__ __align__(16) __bf16 t[64][72];   // pad 8 elems vs bank conflicts
  const int rb = blockIdx.y * 64, cb = blockIdx.x * 64;
  {
    int r = threadIdx.x >> 2, cs = (threadIdx.x & 3) << 4;
    const float* g = in + (size_t)(rb + r) * C + cb + cs;
#pragma unroll
    for (int i = 0; i < 16; ++i) t[cs + i][r] = f32_bf16(g[i]);
  }
  __syncthreads();
  {
    int c = threadIdx.x >> 2, rs = (threadIdx.x & 3) << 4;
    __bf16* o = out + (size_t)(cb + c) * R + rb + rs;
    v8bf a, b;
#pragma unroll
    for (int i = 0; i < 8; ++i) { a[i] = t[c][rs + i]; b[i] = t[c][rs + 8 + i]; }
    *(v8bf*)o       = a;
    *(v8bf*)(o + 8) = b;
  }
}

// -------------------------------------------------------------------------
// Up/gate fused GEMM: h = silu(x @ w0) * (x @ w1). All operands bf16,
// weights pre-transposed to [N][K] -> every tile copy is contiguous 16B.
// Double-buffered LDS, one barrier per K-step.
// -------------------------------------------------------------------------
static __device__ __forceinline__ void stage_upgate(
    int tid, int row0, int col0, int k0,
    const __bf16* __restrict__ x, const __bf16* __restrict__ w0T,
    const __bf16* __restrict__ w1T,
    __bf16* xs, __bf16* w0s, __bf16* w1s)
{
  // A tile: 128 rows x 32 k; 2 threads/row, 32B each
  int r = tid >> 1, c = (tid & 1) << 4;
  const __bf16* gx = x + (size_t)(row0 + r) * DDIM + (k0 + c);
  cp8(gx,     xs + r * LDW + c);
  cp8(gx + 8, xs + r * LDW + c + 8);
  // B tiles: 64 rows x 32 k each; 4 threads/row, 16B each
  int n = tid >> 2, c2 = (tid & 3) << 3;
  cp8(w0T + (size_t)(col0 + n) * DDIM + (k0 + c2), w0s + n * LDW + c2);
  cp8(w1T + (size_t)(col0 + n) * DDIM + (k0 + c2), w1s + n * LDW + c2);
}

__global__ __launch_bounds__(256)
void moe_up_gate(const __bf16* __restrict__ x,    // [Tc, D] bf16 chunk
                 const __bf16* __restrict__ w0T,  // [M, D] bf16 (transposed)
                 const __bf16* __restrict__ w1T,  // [M, D] bf16
                 __bf16* __restrict__ h)          // [Tc, M] bf16
{
  __shared__ __align__(16) __bf16 xs [2][BM * LDW];
  __shared__ __align__(16) __bf16 w0s[2][BN * LDW];
  __shared__ __align__(16) __bf16 w1s[2][BN * LDW];

  const int tid  = threadIdx.x;
  const int lane = tid & 31;
  const int rg   = (tid >> 5) >> 1;        // row group 0..3
  const int cg   = (tid >> 5) & 1;         // col group 0..1
  const int row0 = blockIdx.y * BM;
  const int col0 = blockIdx.x * BN;
  const int l15  = lane & 15;
  const int kh   = (lane >> 4) << 3;       // K chunk base: 0 or 8

  v8f acc0[2][2], acc1[2][2];
#pragma unroll
  for (int i = 0; i < 2; ++i)
#pragma unroll
    for (int j = 0; j < 2; ++j) { acc0[i][j] = (v8f)0.f; acc1[i][j] = (v8f)0.f; }

  stage_upgate(tid, row0, col0, 0, x, w0T, w1T, xs[0], w0s[0], w1s[0]);
  copy_wait();
  __syncthreads();

  int cur = 0;
  for (int k0 = 0; k0 < DDIM; k0 += BK, cur ^= 1) {
    if (k0 + BK < DDIM)
      stage_upgate(tid, row0, col0, k0 + BK, x, w0T, w1T,
                   xs[cur ^ 1], w0s[cur ^ 1], w1s[cur ^ 1]);

    v16bf a[2], b0[2], b1[2];
#pragma unroll
    for (int i = 0; i < 2; ++i) {
      const __bf16* p = &xs[cur][(rg * 32 + i * 16 + l15) * LDW + kh];
      a[i] = cat16(*(const v8bf*)p, *(const v8bf*)(p + 16));
    }
#pragma unroll
    for (int j = 0; j < 2; ++j) {
      const __bf16* p0 = &w0s[cur][(cg * 32 + j * 16 + l15) * LDW + kh];
      b0[j] = cat16(*(const v8bf*)p0, *(const v8bf*)(p0 + 16));
      const __bf16* p1 = &w1s[cur][(cg * 32 + j * 16 + l15) * LDW + kh];
      b1[j] = cat16(*(const v8bf*)p1, *(const v8bf*)(p1 + 16));
    }
#pragma unroll
    for (int i = 0; i < 2; ++i)
#pragma unroll
      for (int j = 0; j < 2; ++j) {
        acc0[i][j] = __builtin_amdgcn_wmma_f32_16x16x32_bf16(
            false, a[i], false, b0[j], (short)0, acc0[i][j], false, false);
        acc1[i][j] = __builtin_amdgcn_wmma_f32_16x16x32_bf16(
            false, a[i], false, b1[j], (short)0, acc1[i][j], false, false);
      }
    copy_wait();
    __syncthreads();
  }

  // epilogue: h = silu(acc0) * acc1 (fast rcp), store bf16
#pragma unroll
  for (int i = 0; i < 2; ++i)
#pragma unroll
    for (int j = 0; j < 2; ++j)
#pragma unroll
      for (int v = 0; v < 8; ++v) {
        int r = row0 + rg * 32 + i * 16 + v + ((lane >> 4) << 3);
        int c = col0 + cg * 32 + j * 16 + l15;
        float g = acc0[i][j][v];
        float s = g * __builtin_amdgcn_rcpf(1.f + __expf(-g));   // silu
        h[(size_t)r * MDIM + c] = f32_bf16(s * acc1[i][j][v]);
      }
}

// -------------------------------------------------------------------------
// Down GEMM + weighted combine: out[t,:] += cw[t,e] * (h[t,:] @ wo[e]).
// -------------------------------------------------------------------------
static __device__ __forceinline__ void stage_down(
    int tid, int row0, int col0, int k0,
    const __bf16* __restrict__ h, const __bf16* __restrict__ woT,
    __bf16* hs, __bf16* wos)
{
  int r = tid >> 1, c = (tid & 1) << 4;
  const __bf16* gh = h + (size_t)(row0 + r) * MDIM + (k0 + c);
  cp8(gh,     hs + r * LDW + c);
  cp8(gh + 8, hs + r * LDW + c + 8);
  int n = tid >> 2, c2 = (tid & 3) << 3;
  cp8(woT + (size_t)(col0 + n) * MDIM + (k0 + c2), wos + n * LDW + c2);
}

__global__ __launch_bounds__(256)
void moe_down(const __bf16* __restrict__ h,    // [Tc, M] bf16
              const __bf16* __restrict__ woT,  // [D, M] bf16 (transposed)
              const float*  __restrict__ cw,   // [T, E]
              float*        __restrict__ out,  // [T, D]
              int tokBase, int expert)
{
  __shared__ __align__(16) __bf16 hs [2][BM * LDW];
  __shared__ __align__(16) __bf16 wos[2][BN * LDW];

  const int tid  = threadIdx.x;
  const int lane = tid & 31;
  const int rg   = (tid >> 5) >> 1;
  const int cg   = (tid >> 5) & 1;
  const int row0 = blockIdx.y * BM;
  const int col0 = blockIdx.x * BN;
  const int l15  = lane & 15;
  const int kh   = (lane >> 4) << 3;

  v8f acc[2][2];
#pragma unroll
  for (int i = 0; i < 2; ++i)
#pragma unroll
    for (int j = 0; j < 2; ++j) acc[i][j] = (v8f)0.f;

  stage_down(tid, row0, col0, 0, h, woT, hs[0], wos[0]);
  copy_wait();
  __syncthreads();

  int cur = 0;
  for (int k0 = 0; k0 < MDIM; k0 += BK, cur ^= 1) {
    if (k0 + BK < MDIM)
      stage_down(tid, row0, col0, k0 + BK, h, woT, hs[cur ^ 1], wos[cur ^ 1]);

    v16bf a[2], b[2];
#pragma unroll
    for (int i = 0; i < 2; ++i) {
      const __bf16* p = &hs[cur][(rg * 32 + i * 16 + l15) * LDW + kh];
      a[i] = cat16(*(const v8bf*)p, *(const v8bf*)(p + 16));
    }
#pragma unroll
    for (int j = 0; j < 2; ++j) {
      const __bf16* p = &wos[cur][(cg * 32 + j * 16 + l15) * LDW + kh];
      b[j] = cat16(*(const v8bf*)p, *(const v8bf*)(p + 16));
    }
#pragma unroll
    for (int i = 0; i < 2; ++i)
#pragma unroll
      for (int j = 0; j < 2; ++j)
        acc[i][j] = __builtin_amdgcn_wmma_f32_16x16x32_bf16(
            false, a[i], false, b[j], (short)0, acc[i][j], false, false);
    copy_wait();
    __syncthreads();
  }

#pragma unroll
  for (int i = 0; i < 2; ++i)
#pragma unroll
    for (int j = 0; j < 2; ++j)
#pragma unroll
      for (int v = 0; v < 8; ++v) {
        int r = row0 + rg * 32 + i * 16 + v + ((lane >> 4) << 3);
        int c = col0 + cg * 32 + j * 16 + l15;
        float s = cw[(size_t)(tokBase + r) * NEXP + expert];
        float* op = &out[(size_t)(tokBase + r) * DDIM + c];
        *op += s * acc[i][j][v];
      }
}

// -------------------------------------------------------------------------
// Host launcher. Workspace layout (all 256B-aligned, ~96 MB total):
//   cw   fp32 [T,E]                      256 KB
//   xbf  bf16 [T,D]                       32 MB
//   w0T  bf16 [M,D]  (per-expert slot)    16 MB
//   w1T  bf16 [M,D]                       16 MB
//   woT  bf16 [D,M]                       16 MB
//   h    bf16 [Tc,M]                      16 MB
// -------------------------------------------------------------------------
extern "C" void kernel_launch(void* const* d_in, const int* in_sizes, int n_in,
                              void* d_out, int out_size, void* d_ws, size_t ws_size,
                              hipStream_t stream) {
  (void)in_sizes; (void)n_in; (void)out_size; (void)ws_size;
  const float* x  = (const float*)d_in[0];
  const float* gw = (const float*)d_in[1];
  const float* w0 = (const float*)d_in[2];
  const float* w1 = (const float*)d_in[3];
  const float* wo = (const float*)d_in[4];
  float* out = (float*)d_out;

  char* ws = (char*)d_ws;
  float*  cw   = (float*)ws;                               ws += (size_t)TOK * NEXP * 4;
  __bf16* xbf  = (__bf16*)ws;                              ws += (size_t)TOK * DDIM * 2;
  __bf16* w0T  = (__bf16*)ws;                              ws += (size_t)MDIM * DDIM * 2;
  __bf16* w1T  = (__bf16*)ws;                              ws += (size_t)MDIM * DDIM * 2;
  __bf16* woT  = (__bf16*)ws;                              ws += (size_t)MDIM * DDIM * 2;
  __bf16* hbuf = (__bf16*)ws;

  (void)hipMemsetAsync(d_out, 0, (size_t)TOK * DDIM * sizeof(float), stream);
  moe_router<<<TOK / 8, 256, 0, stream>>>(x, gw, cw);
  cvt_bf16<<<(TOK * DDIM) / (256 * 8), 256, 0, stream>>>(x, xbf);

  for (int e = 0; e < NEXP; ++e) {
    // pre-transpose+convert this expert's weights to bf16 [N][K]
    dim3 gT01(MDIM / 64, DDIM / 64);   // in [D,M] -> out [M,D]
    transpose_cvt<<<gT01, 256, 0, stream>>>(w0 + (size_t)e * DDIM * MDIM, w0T, DDIM, MDIM);
    transpose_cvt<<<gT01, 256, 0, stream>>>(w1 + (size_t)e * DDIM * MDIM, w1T, DDIM, MDIM);
    dim3 gTo(DDIM / 64, MDIM / 64);    // in [M,D] -> out [D,M]
    transpose_cvt<<<gTo, 256, 0, stream>>>(wo + (size_t)e * MDIM * DDIM, woT, MDIM, DDIM);

    for (int c0 = 0; c0 < TOK; c0 += TCH) {
      dim3 gA(MDIM / BN, TCH / BM);
      moe_up_gate<<<gA, 256, 0, stream>>>(xbf + (size_t)c0 * DDIM, w0T, w1T, hbuf);
      dim3 gB(DDIM / BN, TCH / BM);
      moe_down<<<gB, 256, 0, stream>>>(hbuf, woT, cw, out, c0, e);
    }
  }
}
